// GrowthLayer_52561809769066
// MI455X (gfx1250) — matrix operand order repaired and checked by reading, not compiled, not run on hardware
//
#include <hip/hip_runtime.h>
#include <hip/hip_bf16.h>

// ---------------------------------------------------------------------------
// GrowthLayer for MI455X (gfx1250, wave32):
//   1) V = inputs @ w_in + b_in            (bf16 WMMA GEMM, f32 accum)
//   2) S = [v0, ema(diff(concat(z0, V)))]  (per-channel sequential scan)
//   3) out = S @ w_out + b_out             (bf16 WMMA GEMM, f32 accum)
// ---------------------------------------------------------------------------

typedef __attribute__((ext_vector_type(16))) __bf16 v16bf;
typedef __attribute__((ext_vector_type(2)))  __bf16 v2bf;
typedef __attribute__((ext_vector_type(8)))  float  v8f;

union FragBF {
    v16bf        v;
    unsigned int u[8];
    uint4        q[2];
};

// Pack two f32 -> two bf16 in one u32.
// Prefer the packed-convert builtin; otherwise rely on native __bf16 fptrunc
// (RNE), which gfx1250's bf16 hardware handles directly.
__device__ __forceinline__ unsigned int pack2_bf16(float x, float y) {
#if defined(__has_builtin) && __has_builtin(__builtin_amdgcn_cvt_pk_bf16_f32)
    union { v2bf v; unsigned int u; } cv;
    cv.v = __builtin_amdgcn_cvt_pk_bf16_f32(x, y);
    return cv.u;
#else
    union { v2bf v; unsigned int u; } cv;
    cv.v = (v2bf){(__bf16)x, (__bf16)y};
    return cv.u;
#endif
}

#define BM 128
#define BN 128
#define BK 32
#define LDBK 40   // B tile K-stride (80B): keeps every fragment base 16B-aligned

// A fragment: 16x32 bf16 tile starting at LDS row `row0`.
// Per ISA 7.12.2 (16-bit A 16x32): lanes 0-15 hold row M=lane, K pairs
// {0..7},{16..23}; lanes 16-31 hold the same rows with K offset +8.
// Row stride 64B -> both 16B-chunks are 16B-aligned => ds_load_b128 x2.
__device__ __forceinline__ v16bf load_a_frag(const unsigned short (*As)[BK],
                                             int row0, int lane) {
    const int r    = row0 + (lane & 15);
    const int kadd = (lane & 16) ? 8 : 0;
    const unsigned short* base = &As[r][kadd];
    FragBF f;
    f.q[0] = *(const uint4*)(base);        // K = kadd .. kadd+7
    f.q[1] = *(const uint4*)(base + 16);   // K = kadd+16 .. kadd+23
    return f.v;
}

// B fragment: 32x16 bf16 tile (K x N) for output columns starting at col0.
// Lanes 0-15: column N=lane, K={0..7},{16..23}; lanes 16-31: column N=lane-16, K={8..15},{24..31}.
// Bs stored transposed [n][k], row stride 80B => 16B-aligned ds_load_b128 x2.
__device__ __forceinline__ v16bf load_b_frag(const unsigned short (*Bs)[LDBK],
                                             int col0, int lane) {
    const int n     = col0 + (lane & 15);
    const int kbase = (lane & 16) ? 8 : 0;
    const unsigned short* base = &Bs[n][kbase];
    FragBF f;
    f.q[0] = *(const uint4*)(base);        // K = kbase .. kbase+7
    f.q[1] = *(const uint4*)(base + 16);   // K = kbase+16 .. kbase+23
    return f.v;
}

// C = A(MxK,f32) * B(KxN,f32) + bias, computed in bf16 WMMA with f32 accum.
// Block: 256 threads = 8 waves (4M x 2N); block tile 128x128;
// wave tile 32x64 = 2x4 of 16x16 -> 8 WMMAs per K-step.
// Double-buffered LDS, one barrier per K-step; next tile's global loads are
// issued before the WMMAs so their latency hides under compute.
__global__ __launch_bounds__(256)
void gemm_bias_kernel(const float* __restrict__ A,
                      const float* __restrict__ Bm,
                      const float* __restrict__ bias,
                      float* __restrict__ C,
                      int M, int N, int K) {
    __shared__ unsigned short As[2][BM][BK];    // 16 KB
    __shared__ unsigned short Bs[2][BN][LDBK];  // 20 KB

    const int tid   = threadIdx.x;
    const int lane  = tid & 31;
    const int wave  = tid >> 5;
    const int m_blk = blockIdx.y * BM;
    const int n_blk = blockIdx.x * BN;

    const int wm = (wave >> 1) * 32;   // 0,32,64,96
    const int wn = (wave & 1) * 64;    // 0,64

    // ---- per-thread staging coordinates (loop-invariant) ----
    int ri[4], kci[4], kki[4], nni[4];
    const float4* ap[4];
    const float4* bp[4];
#pragma unroll
    for (int i = 0; i < 4; ++i) {
        const int fa = tid + i * 256;          // A float4 slot 0..1023
        ri[i]  = fa >> 3;                      // 8 float4 per 32-elem row
        kci[i] = (fa & 7) * 4;
        int gr = m_blk + ri[i];
        gr = (gr < M) ? gr : (M - 1);          // clamp tail rows (never stored)
        ap[i] = (const float4*)(A + (size_t)gr * K + kci[i]);

        const int fb = tid + i * 256;          // B float4 slot 0..1023
        kki[i] = fb >> 5;                      // 32 float4 per 128-elem k-row
        nni[i] = (fb & 31) * 4;
        bp[i]  = (const float4*)(Bm + (size_t)kki[i] * N + n_blk + nni[i]);
    }
    const int bstep = (BK * N) / 4;            // float4s per B K-tile

    v8f acc[2][4];
#pragma unroll
    for (int i = 0; i < 2; ++i)
#pragma unroll
        for (int j = 0; j < 4; ++j)
            acc[i][j] = (v8f){0.f, 0.f, 0.f, 0.f, 0.f, 0.f, 0.f, 0.f};

    // ---- prologue: load + stash tile 0 into buffer 0 ----
    float4 ar[4], br[4];
#pragma unroll
    for (int i = 0; i < 4; ++i) { ar[i] = *ap[i]; ap[i] += BK / 4; }
#pragma unroll
    for (int i = 0; i < 4; ++i) { br[i] = *bp[i]; bp[i] += bstep; }
#pragma unroll
    for (int i = 0; i < 4; ++i) {
        *(unsigned int*)&As[0][ri[i]][kci[i] + 0] = pack2_bf16(ar[i].x, ar[i].y);
        *(unsigned int*)&As[0][ri[i]][kci[i] + 2] = pack2_bf16(ar[i].z, ar[i].w);
        const unsigned int p0 = pack2_bf16(br[i].x, br[i].y);
        const unsigned int p1 = pack2_bf16(br[i].z, br[i].w);
        Bs[0][nni[i] + 0][kki[i]] = (unsigned short)(p0 & 0xFFFFu);
        Bs[0][nni[i] + 1][kki[i]] = (unsigned short)(p0 >> 16);
        Bs[0][nni[i] + 2][kki[i]] = (unsigned short)(p1 & 0xFFFFu);
        Bs[0][nni[i] + 3][kki[i]] = (unsigned short)(p1 >> 16);
    }
    __syncthreads();

    const int ktiles = K / BK;
    for (int kt = 0; kt < ktiles; ++kt) {
        const int  cur      = kt & 1;
        const int  nxt      = cur ^ 1;
        const bool havenext = (kt + 1) < ktiles;

        // 1) issue next tile's global loads (latency hides under WMMAs)
        if (havenext) {
#pragma unroll
            for (int i = 0; i < 4; ++i) { ar[i] = *ap[i]; ap[i] += BK / 4; }
#pragma unroll
            for (int i = 0; i < 4; ++i) { br[i] = *bp[i]; bp[i] += bstep; }
        }
        // speculative prefetch of the tile after that
        if (kt + 2 < ktiles)
            __builtin_prefetch(ap[0], 0, 1);

        // 2) fragment loads + 8 WMMAs from current buffer
        const v16bf a0 = load_a_frag(As[cur], wm,      lane);
        const v16bf a1 = load_a_frag(As[cur], wm + 16, lane);
        v16bf bf[4];
#pragma unroll
        for (int j = 0; j < 4; ++j)
            bf[j] = load_b_frag(Bs[cur], wn + j * 16, lane);

#pragma unroll
        for (int j = 0; j < 4; ++j) {
            acc[0][j] = __builtin_amdgcn_wmma_f32_16x16x32_bf16(
                false, a0, false, bf[j], (short)0, acc[0][j], false, false);
            acc[1][j] = __builtin_amdgcn_wmma_f32_16x16x32_bf16(
                false, a1, false, bf[j], (short)0, acc[1][j], false, false);
        }

        // 3) convert + stash next tile into the other buffer
        if (havenext) {
#pragma unroll
            for (int i = 0; i < 4; ++i) {
                *(unsigned int*)&As[nxt][ri[i]][kci[i] + 0] =
                    pack2_bf16(ar[i].x, ar[i].y);
                *(unsigned int*)&As[nxt][ri[i]][kci[i] + 2] =
                    pack2_bf16(ar[i].z, ar[i].w);
                const unsigned int p0 = pack2_bf16(br[i].x, br[i].y);
                const unsigned int p1 = pack2_bf16(br[i].z, br[i].w);
                Bs[nxt][nni[i] + 0][kki[i]] = (unsigned short)(p0 & 0xFFFFu);
                Bs[nxt][nni[i] + 1][kki[i]] = (unsigned short)(p0 >> 16);
                Bs[nxt][nni[i] + 2][kki[i]] = (unsigned short)(p1 & 0xFFFFu);
                Bs[nxt][nni[i] + 3][kki[i]] = (unsigned short)(p1 >> 16);
            }
        }
        __syncthreads();
    }

    // ---- store: C/D 16x16 f32 layout (8 VGPRs); add bias ----
    const int col_in  = lane & 15;
    const int row_off = (lane & 16) ? 8 : 0;
#pragma unroll
    for (int i = 0; i < 2; ++i) {
#pragma unroll
        for (int j = 0; j < 4; ++j) {
            const int col = n_blk + wn + j * 16 + col_in;
            const float bv = bias[col];
#pragma unroll
            for (int r = 0; r < 8; ++r) {
                const int row = m_blk + wm + i * 16 + row_off + r;
                if (row < M)
                    C[(size_t)row * N + col] = acc[i][j][r] + bv;
            }
        }
    }
}

// Per-channel temporal diff + exponential smoothing.
// S[b][0][c] = v0[c];  S[b][t+1][c] = a*h + (1-a)*(V[b][t][c]-V[b][t-1][c])
// Unrolled x4 so four independent loads are in flight per chain step.
__global__ __launch_bounds__(256)
void scan_kernel(const float* __restrict__ V,
                 const float* __restrict__ z0,
                 const float* __restrict__ v0,
                 const float* __restrict__ logit,
                 float* __restrict__ S,
                 int B, int T) {
    const int Cc  = 512;                     // H*Dh
    const int tid = blockIdx.x * blockDim.x + threadIdx.x;
    if (tid >= B * Cc) return;
    const int b = tid / Cc;
    const int c = tid % Cc;
    const int h = c >> 6;                    // Dh = 64

    const float a   = 1.f / (1.f + __expf(-logit[h]));
    const float oma = 1.f - a;

    float prev = z0[c];
    float hs   = v0[c];

    const float* vp = V + (size_t)b * T * Cc + c;
    float*       sp = S + (size_t)b * (T + 1) * Cc + c;

    sp[0] = v0[c];
    for (int t = 0; t < T; t += 4) {
        const float x0 = vp[(size_t)(t + 0) * Cc];
        const float x1 = vp[(size_t)(t + 1) * Cc];
        const float x2 = vp[(size_t)(t + 2) * Cc];
        const float x3 = vp[(size_t)(t + 3) * Cc];

        hs = a * hs + oma * (x0 - prev);
        sp[(size_t)(t + 1) * Cc] = hs;
        hs = a * hs + oma * (x1 - x0);
        sp[(size_t)(t + 2) * Cc] = hs;
        hs = a * hs + oma * (x2 - x1);
        sp[(size_t)(t + 3) * Cc] = hs;
        hs = a * hs + oma * (x3 - x2);
        sp[(size_t)(t + 4) * Cc] = hs;
        prev = x3;
    }
}

extern "C" void kernel_launch(void* const* d_in, const int* in_sizes, int n_in,
                              void* d_out, int out_size, void* d_ws, size_t ws_size,
                              hipStream_t stream) {
    (void)in_sizes; (void)n_in; (void)out_size; (void)ws_size;

    const float* inputs = (const float*)d_in[0];  // (B,T,D)
    const float* z0     = (const float*)d_in[1];  // (H,Dh)
    const float* w_in   = (const float*)d_in[2];  // (D, H*Dh)
    const float* b_in   = (const float*)d_in[3];  // (H*Dh)
    const float* v0     = (const float*)d_in[4];  // (H,Dh)
    const float* logit  = (const float*)d_in[5];  // (H,1)
    const float* w_out  = (const float*)d_in[6];  // (H*Dh, D)
    const float* b_out  = (const float*)d_in[7];  // (D)
    float*       out    = (float*)d_out;          // (B, T+1, D)

    const int B = 16, T = 2048, D = 512, C = 512; // C = H*Dh

    float* V = (float*)d_ws;                                 // B*T*C
    float* S = V + (size_t)B * T * C;                        // B*(T+1)*C

    // 1) V = inputs @ w_in + b_in    (M = B*T)
    {
        const int M = B * T;
        dim3 grid(C / BN, (M + BM - 1) / BM);
        gemm_bias_kernel<<<grid, 256, 0, stream>>>(inputs, w_in, b_in, V,
                                                   M, C, D);
    }

    // 2) scan -> S (B, T+1, C)
    {
        const int threads = B * C;
        scan_kernel<<<(threads + 255) / 256, 256, 0, stream>>>(V, z0, v0,
                                                               logit, S, B, T);
    }

    // 3) out = S @ w_out + b_out    (M = B*(T+1))
    {
        const int M = B * (T + 1);
        dim3 grid(D / BN, (M + BM - 1) / BM);
        gemm_bias_kernel<<<grid, 256, 0, stream>>>(S, w_out, b_out, out,
                                                   M, D, C);
    }
}
